// CS224NRnnModel_66468913873179
// MI455X (gfx1250) — compile-verified, matrix operand order checked by reading, not run on hardware
//
#include <hip/hip_runtime.h>
#include <hip/hip_bf16.h>

typedef __attribute__((ext_vector_type(16))) __bf16 v16bf;
typedef __attribute__((ext_vector_type(8)))  __bf16 v8bf;
typedef __attribute__((ext_vector_type(8)))  float  v8f;

#define NV_  32000
#define NE_  1024
#define NH_  1024
#define NB_  32
#define NT_  128

// ---------- WMMA helpers (CDNA5 gfx1250, wave32) ----------
// 16-bit A/B fragment: lane (half = lane>>4, r = lane&15) holds row/col r,
// K elements [8*half, 8*half+8) in vector elems 0..7 and
// K elements [16+8*half, 16+8*half+8) in vector elems 8..15.
static __device__ __forceinline__ v16bf frag16(const __bf16* base) {
  v8bf lo = *(const v8bf*)(base);
  v8bf hi = *(const v8bf*)(base + 16);
  return __builtin_shufflevector(lo, hi, 0,1,2,3,4,5,6,7,8,9,10,11,12,13,14,15);
}

static __device__ __forceinline__ v8f wmma_bf16(v16bf a, v16bf b, v8f c) {
  // D = A(16x32) * B(32x16) + C, f32 accumulate
  return __builtin_amdgcn_wmma_f32_16x16x32_bf16(false, a, false, b, (short)0, c,
                                                 false, false);
}

// Async copy of 16 bytes/lane from global to LDS (tracked by ASYNCcnt).
// saddr form: addr = SGPR64 base + VGPR32 byte offset.
static __device__ __forceinline__ void async_ld128(unsigned lds_byte_off,
                                                   const void* gbase,
                                                   unsigned gbyte_off) {
  unsigned long long b = (unsigned long long)gbase;
  asm volatile("global_load_async_to_lds_b128 %0, %1, %2"
               :: "v"(lds_byte_off), "v"(gbyte_off), "s"(b)
               : "memory");
}

// ---------- elementwise / data movement kernels ----------
__global__ void cvt_f32_bf16(const float* __restrict__ s, __bf16* __restrict__ d, long n) {
  long i = (long)blockIdx.x * blockDim.x + threadIdx.x;
  long stride = (long)gridDim.x * blockDim.x;
  for (; i < n; i += stride) d[i] = (__bf16)s[i];
}

__global__ void copy_f32(const float* __restrict__ s, float* __restrict__ d, long n) {
  long i = (long)blockIdx.x * blockDim.x + threadIdx.x;
  long stride = (long)gridDim.x * blockDim.x;
  for (; i < n; i += stride) d[i] = s[i];
}

// x_bf16[row, :] = bf16(emb[input_x[row], :]), row = b*T + t
__global__ void embed_gather(const int* __restrict__ idx, const float* __restrict__ emb,
                             __bf16* __restrict__ x) {
  const int row = blockIdx.x;
  const int tok = idx[row];
  const float* __restrict__ src = emb + (size_t)tok * NE_;
  __bf16* __restrict__ dst = x + (size_t)row * NE_;
  for (int i = threadIdx.x; i < NE_; i += blockDim.x) dst[i] = (__bf16)src[i];
}

// last_output[b, :] = output_prob[b*T + (T-1), :]
__global__ void copy_last_rows(const float* __restrict__ outprob, float* __restrict__ dst) {
  const int b = blockIdx.y;
  const int i = blockIdx.x * blockDim.x + threadIdx.x;
  dst[(size_t)b * NV_ + i] = outprob[((size_t)b * NT_ + (NT_ - 1)) * NV_ + i];
}

// ---------- tiled WMMA GEMM: C[M,N] = A[M,K](bf16) * Bw[N,K]^T(bf16) + bias ----------
// grid = (N/128, M/128), block = 256 (8 waves). Wave (wm,wn) computes 64x32.
// Double-buffered async global->LDS staging; NT stores for the f32 output stream.
#define BK 32
#define LDSROW 40           // 32 data + 8 pad bf16 -> 80B row stride, conflict-free b128
#define TILE_ELEMS (128 * LDSROW)          // 5120 bf16 per A or B tile
#define BUF_ELEMS  (2 * TILE_ELEMS)        // A + B per buffer
#define SMEM_BYTES (2 * BUF_ELEMS * 2)     // two buffers, bf16

extern __shared__ __bf16 smem[];

__global__ __launch_bounds__(256)
void gemm_bf16_bias(const __bf16* __restrict__ A, int lda,
                    const __bf16* __restrict__ Bw, int ldb,
                    const float* __restrict__ bias,
                    float* __restrict__ C, int ldc, int K) {
  const int tid  = threadIdx.x;
  const int lane = tid & 31;
  const int wave = tid >> 5;        // 0..7
  const int wm   = wave >> 2;       // 0..1 -> 64-row strip
  const int wn   = wave & 3;        // 0..3 -> 32-col strip
  const int half = lane >> 4;
  const int r    = lane & 15;

  const unsigned m0 = blockIdx.y * 128;
  const unsigned n0 = blockIdx.x * 128;

  // Per-thread staging: two 16B chunks per tile (8KB tile / 256 threads / 16B = 2).
  const unsigned srow = tid >> 2;           // 0..63 (and +64)
  const unsigned scol = (tid & 3) * 8;      // element column: 0,8,16,24

  v8f acc[4][2] = {};

  // stage K-block into buffer `buf` (4 async b128 per thread => ASYNCcnt += 4/wave)
  auto stage = [&](int buf, int k0) {
    const unsigned abase = (unsigned)buf * BUF_ELEMS;
    const unsigned bbase = abase + TILE_ELEMS;
    unsigned ldsA = (abase + srow * LDSROW + scol) * 2u;
    unsigned ga   = ((m0 + srow) * (unsigned)lda + (unsigned)k0 + scol) * 2u;
    async_ld128(ldsA, A, ga);
    async_ld128(ldsA + 64u * LDSROW * 2u, A, ga + 64u * (unsigned)lda * 2u);
    unsigned ldsB = (bbase + srow * LDSROW + scol) * 2u;
    unsigned gb   = ((n0 + srow) * (unsigned)ldb + (unsigned)k0 + scol) * 2u;
    async_ld128(ldsB, Bw, gb);
    async_ld128(ldsB + 64u * LDSROW * 2u, Bw, gb + 64u * (unsigned)ldb * 2u);
  };

  stage(0, 0);
  for (int k0 = 0, it = 0; k0 < K; k0 += BK, ++it) {
    const int cur = it & 1;
    const bool hasNext = (k0 + BK) < K;
    if (hasNext) {
      stage(cur ^ 1, k0 + BK);
      asm volatile("s_wait_asynccnt 0x4" ::: "memory");   // current buffer's 4 done
    } else {
      asm volatile("s_wait_asynccnt 0x0" ::: "memory");
    }
    __syncthreads();  // all waves' async writes to buf[cur] visible

    const __bf16* As = smem + (unsigned)cur * BUF_ELEMS;
    const __bf16* Bs = As + TILE_ELEMS;

    v16bf af[4], bf[2];
#pragma unroll
    for (int mt = 0; mt < 4; ++mt)
      af[mt] = frag16(As + (wm * 64 + mt * 16 + r) * LDSROW + 8 * half);
#pragma unroll
    for (int nt = 0; nt < 2; ++nt)
      bf[nt] = frag16(Bs + (wn * 32 + nt * 16 + r) * LDSROW + 8 * half);
#pragma unroll
    for (int mt = 0; mt < 4; ++mt)
#pragma unroll
      for (int nt = 0; nt < 2; ++nt)
        acc[mt][nt] = wmma_bf16(af[mt], bf[nt], acc[mt][nt]);

    __syncthreads();  // all waves done reading buf[cur] before it is re-staged
  }

#pragma unroll
  for (int nt = 0; nt < 2; ++nt) {
    const int n = n0 + wn * 32 + nt * 16 + r;
    const float bv = bias[n];
#pragma unroll
    for (int mt = 0; mt < 4; ++mt) {
#pragma unroll
      for (int rr = 0; rr < 8; ++rr) {
        const int m = m0 + wm * 64 + mt * 16 + rr + 8 * half;
        __builtin_nontemporal_store(acc[mt][nt][rr] + bv, &C[(size_t)m * ldc + n]);
      }
    }
  }
}

// ---------- one recurrent timestep: h = tanh(xw_t + h_prev @ W_hh^T + b_hh) ----------
// M=32(batch), N=1024, K=1024. grid = 16 (64 cols each), block = 128 (4 waves, 16 cols/wave).
__global__ __launch_bounds__(128)
void rnn_step(const __bf16* __restrict__ hsrc, long ldh,
              const __bf16* __restrict__ W,           // [1024,1024] row-major (n,k)
              const float* __restrict__ xw, long ldx,  // xw + b*ldx + n
              const float* __restrict__ bias,
              __bf16* __restrict__ ysout, long ldy,    // ysout + b*ldy + n
              float* __restrict__ hlast) {
  const int tid  = threadIdx.x;
  const int lane = tid & 31;
  const int wave = tid >> 5;     // 0..3
  const int half = lane >> 4;
  const int r    = lane & 15;
  const int n0   = blockIdx.x * 64 + wave * 16;

  v8f acc[2] = {};
  for (int k0 = 0; k0 < NH_; k0 += 32) {
    v16bf a0 = frag16(hsrc + (long)(0 * 16 + r) * ldh + k0 + 8 * half);
    v16bf a1 = frag16(hsrc + (long)(1 * 16 + r) * ldh + k0 + 8 * half);
    v16bf bw = frag16(W + (long)(n0 + r) * NH_ + k0 + 8 * half);
    acc[0] = wmma_bf16(a0, bw, acc[0]);
    acc[1] = wmma_bf16(a1, bw, acc[1]);
  }

  const int n = n0 + r;
  const float bv = bias[n];
#pragma unroll
  for (int mt = 0; mt < 2; ++mt) {
#pragma unroll
    for (int rr = 0; rr < 8; ++rr) {
      const int b = mt * 16 + rr + 8 * half;     // batch index 0..31
      float v = acc[mt][rr] + xw[(long)b * ldx + n] + bv;
      float h = tanhf(v);
      ysout[(long)b * ldy + n] = (__bf16)h;
      hlast[(long)b * NH_ + n] = h;
    }
  }
}

// ---------- host orchestration ----------
extern "C" void kernel_launch(void* const* d_in, const int* in_sizes, int n_in,
                              void* d_out, int out_size, void* d_ws, size_t ws_size,
                              hipStream_t stream) {
  (void)in_sizes; (void)n_in; (void)out_size; (void)ws_size;
  const int*   input_x = (const int*)d_in[0];
  const float* hidden  = (const float*)d_in[1];
  const float* emb     = (const float*)d_in[2];
  const float* W_ih0   = (const float*)d_in[3];
  const float* W_hh0   = (const float*)d_in[4];
  const float* b_ih0   = (const float*)d_in[5];
  const float* b_hh0   = (const float*)d_in[6];
  const float* W_ih1   = (const float*)d_in[7];
  const float* W_hh1   = (const float*)d_in[8];
  const float* b_ih1   = (const float*)d_in[9];
  const float* b_hh1   = (const float*)d_in[10];
  const float* W_out   = (const float*)d_in[11];
  const float* b_out   = (const float*)d_in[12];
  float* out = (float*)d_out;

  const long BT = (long)NB_ * NT_;             // 4096
  const long HH = (long)NH_ * NH_;             // 1M

  // workspace layout
  char* p = (char*)d_ws;
  auto alloc = [&](size_t bytes) { char* q = p; p += (bytes + 255) & ~(size_t)255; return q; };
  __bf16* wih0b = (__bf16*)alloc(HH * 2);
  __bf16* whh0b = (__bf16*)alloc(HH * 2);
  __bf16* wih1b = (__bf16*)alloc(HH * 2);
  __bf16* whh1b = (__bf16*)alloc(HH * 2);
  __bf16* woutb = (__bf16*)alloc((size_t)NV_ * NH_ * 2);
  __bf16* xb    = (__bf16*)alloc((size_t)BT * NE_ * 2);
  float*  xw    = (float*) alloc((size_t)BT * NH_ * 4);   // reused for both layers
  __bf16* ys0b  = (__bf16*)alloc((size_t)BT * NH_ * 2);
  __bf16* ys1b  = (__bf16*)alloc((size_t)BT * NH_ * 2);
  __bf16* h0b   = (__bf16*)alloc((size_t)NB_ * NH_ * 2);
  __bf16* h1b   = (__bf16*)alloc((size_t)NB_ * NH_ * 2);
  float*  hl0   = (float*) alloc((size_t)NB_ * NH_ * 4);
  float*  hl1   = (float*) alloc((size_t)NB_ * NH_ * 4);

  // 1) weight / hidden conversions to bf16
  cvt_f32_bf16<<<1024, 256, 0, stream>>>(W_ih0, wih0b, HH);
  cvt_f32_bf16<<<1024, 256, 0, stream>>>(W_hh0, whh0b, HH);
  cvt_f32_bf16<<<1024, 256, 0, stream>>>(W_ih1, wih1b, HH);
  cvt_f32_bf16<<<1024, 256, 0, stream>>>(W_hh1, whh1b, HH);
  cvt_f32_bf16<<<4096, 256, 0, stream>>>(W_out, woutb, (long)NV_ * NH_);
  cvt_f32_bf16<<<128, 256, 0, stream>>>(hidden,             h0b, (long)NB_ * NH_);
  cvt_f32_bf16<<<128, 256, 0, stream>>>(hidden + NB_ * NH_, h1b, (long)NB_ * NH_);

  // 2) embedding gather (+ bf16 downconvert)
  embed_gather<<<(int)BT, 256, 0, stream>>>(input_x, emb, xb);

  // 3) layer 0 input projection: xw = x @ W_ih0^T + b_ih0   [4096,1024]
  gemm_bf16_bias<<<dim3(NH_ / 128, (int)(BT / 128)), 256, SMEM_BYTES, stream>>>(
      xb, NE_, wih0b, NE_, b_ih0, xw, NH_, NE_);

  // 4) layer 0 recurrent scan
  for (int t = 0; t < NT_; ++t) {
    const __bf16* hsrc = (t == 0) ? h0b : (ys0b + (size_t)(t - 1) * NH_);
    long ldh = (t == 0) ? (long)NH_ : (long)NT_ * NH_;
    rnn_step<<<dim3(NH_ / 64), 128, 0, stream>>>(
        hsrc, ldh, whh0b, xw + (size_t)t * NH_, (long)NT_ * NH_,
        b_hh0, ys0b + (size_t)t * NH_, (long)NT_ * NH_, hl0);
  }

  // 5) layer 1 input projection: xw = ys0 @ W_ih1^T + b_ih1
  gemm_bf16_bias<<<dim3(NH_ / 128, (int)(BT / 128)), 256, SMEM_BYTES, stream>>>(
      ys0b, NH_, wih1b, NH_, b_ih1, xw, NH_, NH_);

  // 6) layer 1 recurrent scan
  for (int t = 0; t < NT_; ++t) {
    const __bf16* hsrc = (t == 0) ? h1b : (ys1b + (size_t)(t - 1) * NH_);
    long ldh = (t == 0) ? (long)NH_ : (long)NT_ * NH_;
    rnn_step<<<dim3(NH_ / 64), 128, 0, stream>>>(
        hsrc, ldh, whh1b, xw + (size_t)t * NH_, (long)NT_ * NH_,
        b_hh1, ys1b + (size_t)t * NH_, (long)NT_ * NH_, hl1);
  }

  // 7) output projection: out[4096,32000] = ys1 @ W_out^T + b_out
  gemm_bf16_bias<<<dim3(NV_ / 128, (int)(BT / 128)), 256, SMEM_BYTES, stream>>>(
      ys1b, NH_, woutb, NH_, b_out, out, NV_, NH_);

  // 8) rnn_hidden [2,32,1024] and last_output [32,32000]
  float* out_hidden = out + (size_t)BT * NV_;
  float* out_last   = out_hidden + 2L * NB_ * NH_;
  copy_f32<<<128, 256, 0, stream>>>(hl0, out_hidden, (long)NB_ * NH_);
  copy_f32<<<128, 256, 0, stream>>>(hl1, out_hidden + NB_ * NH_, (long)NB_ * NH_);
  copy_last_rows<<<dim3(NV_ / 256, NB_), 256, 0, stream>>>(out, out_last);
}
